// SAT_25898652795473
// MI455X (gfx1250) — matrix operand order repaired and checked by reading, not compiled, not run on hardware
//
#include <hip/hip_runtime.h>
#include <hip/hip_bf16.h>
#include <stdint.h>
#include <math.h>

typedef __attribute__((ext_vector_type(16))) __bf16 v16bf;
typedef __attribute__((ext_vector_type(8)))  float  v8f;
typedef __attribute__((__vector_size__(16))) int    v4i;
typedef unsigned short u16;
typedef unsigned int   u32;

#define DEV __device__ __forceinline__

DEV float bf2f(u16 b){ u32 u = ((u32)b) << 16; float f; __builtin_memcpy(&f, &u, 4); return f; }
DEV u16 f2bf(float f){
  u32 u; __builtin_memcpy(&u, &f, 4);
  u32 r = u + 0x7FFFu + ((u >> 16) & 1u);
  return (u16)(r >> 16);
}

union FragB { v16bf v; uint4 q[2]; };

DEV v8f zero8(){
  v8f z;
#pragma unroll
  for(int i = 0; i < 8; i++) z[i] = 0.f;
  return z;
}

// ---- gfx1250 async global->LDS (confirmed present; param0 = AS1 v4i*) ----
#if defined(__has_builtin)
#if __has_builtin(__builtin_amdgcn_global_load_async_to_lds_b128)
#define HAVE_ASYNC_LDS 1
#endif
#endif
#ifndef HAVE_ASYNC_LDS
#define HAVE_ASYNC_LDS 0
#endif

#define AS1_V4I(p) ((__attribute__((address_space(1))) v4i*)(unsigned long long)(p))
#define AS3_V4I(p) ((__attribute__((address_space(3))) v4i*)(p))

DEV void async_b128(const u16* g, u16* l){
#if HAVE_ASYNC_LDS
  __builtin_amdgcn_global_load_async_to_lds_b128(AS1_V4I(g), AS3_V4I(l), 0, 0);
#else
  (void)g; (void)l;
#endif
}
DEV void wait_async0(){
#if HAVE_ASYNC_LDS
#if __has_builtin(__builtin_amdgcn_s_wait_asynccnt)
  __builtin_amdgcn_s_wait_asynccnt(0);
#else
  asm volatile("s_wait_asynccnt 0" ::: "memory");
#endif
#endif
}

// ---------------------------------------------------------------------------
// f32 [R][C] -> bf16 transposed [C][R]
// ---------------------------------------------------------------------------
__global__ void k_cvt_transpose(const float* __restrict__ in, u16* __restrict__ out,
                                int R, int C){
  __shared__ u16 t[32][33];
  int c0 = blockIdx.x * 32, r0 = blockIdx.y * 32;
  int tx = threadIdx.x & 31, ty = threadIdx.x >> 5;
  for(int i = ty; i < 32; i += 8) t[i][tx] = f2bf(in[(long)(r0 + i) * C + c0 + tx]);
  __syncthreads();
  for(int i = ty; i < 32; i += 8) out[(long)(c0 + i) * R + r0 + tx] = t[tx][i];
}

// ---------------------------------------------------------------------------
// Row LayerNorm, optional residual add; outputs bf16 and/or f32.
// ---------------------------------------------------------------------------
template<int ADD, int OBF, int OF32>
__global__ void k_layernorm(const float* __restrict__ x, const float* __restrict__ r,
                            const float* __restrict__ g, const float* __restrict__ bta,
                            u16* __restrict__ ob, float* __restrict__ of, int N){
  int row = blockIdx.x;
  const float* xr = x + (long)row * N;
  const float* rr = r + (long)row * N;
  __shared__ float s1[256], s2[256];
  float lsum = 0.f, lsq = 0.f;
  for(int i = threadIdx.x; i < N; i += 256){
    float v = xr[i];
    if(ADD) v += rr[i];
    lsum += v; lsq += v * v;
  }
  s1[threadIdx.x] = lsum; s2[threadIdx.x] = lsq;
  __syncthreads();
  for(int s = 128; s > 0; s >>= 1){
    if(threadIdx.x < s){ s1[threadIdx.x] += s1[threadIdx.x + s]; s2[threadIdx.x] += s2[threadIdx.x + s]; }
    __syncthreads();
  }
  float mean = s1[0] / N;
  float var  = s2[0] / N - mean * mean;
  float rstd = rsqrtf(var + 1e-5f);
  for(int i = threadIdx.x; i < N; i += 256){
    float v = xr[i];
    if(ADD) v += rr[i];
    float o = (v - mean) * rstd * g[i] + bta[i];
    if(OBF)  ob[(long)row * N + i] = f2bf(o);
    if(OF32) of[(long)row * N + i] = o;
  }
}

__global__ void k_init_slots(const float* __restrict__ noise, const float* __restrict__ mu,
                             const float* __restrict__ ls, float* __restrict__ slots,
                             int total, int Dv){
  int i = blockIdx.x * 256 + threadIdx.x;
  if(i < total){
    int d = i % Dv;
    slots[i] = mu[d] + expf(ls[d]) * noise[i];
  }
}

// ---------------------------------------------------------------------------
// Big GEMM: C[M,N] = A[M,K] @ B[K,N] with B pre-transposed: BT[N][K].
// bf16 in, f32 WMMA accumulate. Block tile 128x64, BK=64, double-buffered LDS
// with async global->LDS prefetch. 8 waves (4x2), 8 WMMA per k-step.
// OMODE: 0 bf16 store (LDS-staged, coalesced b128)
//        1 f32 store
//        2 f32 += (residual, in place)
//        3 bf16 store transposed within 1024-row batches: out[b][n][m%1024]
// Requires M%128==0, N%64==0, K%64==0.
// ---------------------------------------------------------------------------
template<int OMODE>
__global__ __launch_bounds__(256)
void k_gemm(const u16* __restrict__ A, const u16* __restrict__ BT,
            u16* __restrict__ Cb, float* __restrict__ Cf,
            int M, int N, int K){
  __shared__ u16 Al[2][128][72];
  __shared__ u16 Bl[2][64][72];
  const int tid  = threadIdx.x;
  const int lane = tid & 31, w = tid >> 5;
  const int wm = w >> 1, wn = w & 1;
  const int m0 = blockIdx.y * 128, n0 = blockIdx.x * 64;
  const int mloc = lane & 15, kh = lane >> 4;

  const int arow = tid >> 1, acs = (tid & 1) * 32;   // A: 128 rows x 2 x (4 uint4)
  const int brow = tid >> 2, bcs = (tid & 3) * 16;   // B: 64 rows x 4 x (2 uint4)

  v8f acc[2][2];
  acc[0][0] = zero8(); acc[0][1] = zero8(); acc[1][0] = zero8(); acc[1][1] = zero8();

  uint4 ar[4], br[2];
  auto gload = [&](int k0){
#pragma unroll
    for(int j = 0; j < 4; j++) ar[j] = *(const uint4*)&A[(long)(m0 + arow) * K + k0 + acs + j * 8];
#pragma unroll
    for(int j = 0; j < 2; j++) br[j] = *(const uint4*)&BT[(long)(n0 + brow) * K + k0 + bcs + j * 8];
  };
  auto lstore = [&](int buf){
#pragma unroll
    for(int j = 0; j < 4; j++) *(uint4*)&Al[buf][arow][acs + j * 8] = ar[j];
#pragma unroll
    for(int j = 0; j < 2; j++) *(uint4*)&Bl[buf][brow][bcs + j * 8] = br[j];
  };
  auto gasync = [&](int k0, int buf){
#pragma unroll
    for(int j = 0; j < 4; j++)
      async_b128(&A[(long)(m0 + arow) * K + k0 + acs + j * 8], &Al[buf][arow][acs + j * 8]);
#pragma unroll
    for(int j = 0; j < 2; j++)
      async_b128(&BT[(long)(n0 + brow) * K + k0 + bcs + j * 8], &Bl[buf][brow][bcs + j * 8]);
  };
  auto compute = [&](int buf){
#pragma unroll
    for(int q = 0; q < 2; q++){
      FragB af[2], bf_[2];
#pragma unroll
      for(int mt = 0; mt < 2; mt++){
        int rr = wm * 32 + mt * 16 + mloc;
        af[mt].q[0] = *(const uint4*)&Al[buf][rr][q * 32 + kh * 16];
        af[mt].q[1] = *(const uint4*)&Al[buf][rr][q * 32 + kh * 16 + 8];
      }
#pragma unroll
      for(int nt = 0; nt < 2; nt++){
        int cc = wn * 32 + nt * 16 + mloc;
        bf_[nt].q[0] = *(const uint4*)&Bl[buf][cc][q * 32 + kh * 16];
        bf_[nt].q[1] = *(const uint4*)&Bl[buf][cc][q * 32 + kh * 16 + 8];
      }
#pragma unroll
      for(int mt = 0; mt < 2; mt++)
#pragma unroll
        for(int nt = 0; nt < 2; nt++)
          acc[mt][nt] = __builtin_amdgcn_wmma_f32_16x16x32_bf16(
              false, af[mt].v, false, bf_[nt].v, (short)0, acc[mt][nt], false, false);
    }
  };

  const int nk = K >> 6;
#if HAVE_ASYNC_LDS
  gasync(0, 0);
  wait_async0();
  __syncthreads();
  for(int kt = 0; kt < nk; kt++){
    int buf = kt & 1;
    if(kt + 1 < nk) gasync((kt + 1) * 64, buf ^ 1);
    compute(buf);
    wait_async0();
    __syncthreads();
  }
#else
  gload(0);
  lstore(0);
  __syncthreads();
  for(int kt = 0; kt < nk; kt++){
    int buf = kt & 1;
    if(kt + 1 < nk) gload((kt + 1) * 64);
    compute(buf);
    if(kt + 1 < nk) lstore(buf ^ 1);
    __syncthreads();
  }
#endif

  if(OMODE == 0){
    // stage bf16 C tile in LDS, then coalesced b128 stores
    u16 (*Cs)[72] = (u16 (*)[72])&Al[0][0][0];
#pragma unroll
    for(int mt = 0; mt < 2; mt++)
#pragma unroll
      for(int nt = 0; nt < 2; nt++)
#pragma unroll
        for(int r = 0; r < 8; r++)
          Cs[wm * 32 + mt * 16 + r + 8 * kh][wn * 32 + nt * 16 + mloc] = f2bf(acc[mt][nt][r]);
    __syncthreads();
    const int row = tid >> 1, cs2 = (tid & 1) * 32;
#pragma unroll
    for(int j = 0; j < 4; j++)
      *(uint4*)&Cb[(long)(m0 + row) * N + n0 + cs2 + j * 8] = *(uint4*)&Cs[row][cs2 + j * 8];
  } else if(OMODE == 3){
    // bf16, transposed within 1024-row batches: out[b][n][j], 8 contiguous j per lane
#pragma unroll
    for(int mt = 0; mt < 2; mt++)
#pragma unroll
      for(int nt = 0; nt < 2; nt++){
        u16 tmp[8];
#pragma unroll
        for(int r = 0; r < 8; r++) tmp[r] = f2bf(acc[mt][nt][r]);
        long mi0 = m0 + wm * 32 + mt * 16 + 8 * kh;
        long bidx = mi0 >> 10;
        long j0 = mi0 & 1023;
        long n = n0 + wn * 32 + nt * 16 + mloc;
        *(uint4*)&Cb[bidx * (long)N * 1024 + n * 1024 + j0] = *(uint4*)tmp;
      }
  } else {
#pragma unroll
    for(int mt = 0; mt < 2; mt++)
#pragma unroll
      for(int nt = 0; nt < 2; nt++){
        int n = n0 + wn * 32 + nt * 16 + mloc;
#pragma unroll
        for(int r = 0; r < 8; r++){
          long mi  = m0 + wm * 32 + mt * 16 + r + 8 * kh;
          long idx = mi * N + n;
          if(OMODE == 1) Cf[idx] = acc[mt][nt][r];
          else           Cf[idx] += acc[mt][nt][r];
        }
      }
  }
}

// ---------------------------------------------------------------------------
// Batched M=16 GEMM: C[b][16][N] = alpha * A[b][16][K] @ B[b]^T
// B supplied as [N][K] rows per batch (k natively; v pre-transposed).
// Block tile 16x128, BK=64, double-buffered async LDS, 2 WMMA per step/wave.
// f32 output, LDS-staged coalesced stores.
// ---------------------------------------------------------------------------
__global__ __launch_bounds__(256)
void k_gemm16(const u16* __restrict__ Abase, const u16* __restrict__ Bbase,
              float* __restrict__ Cf, int N, int K, float alpha){
  const int b  = blockIdx.y;
  const int n0 = blockIdx.x * 128;
  const u16* A  = Abase + (long)b * 16 * K;
  const u16* Bm = Bbase + (long)b * (long)N * K;

  __shared__ u16 Al[2][16][72];
  __shared__ u16 Bl[2][128][72];
  const int tid = threadIdx.x, lane = tid & 31, w = tid >> 5;
  const int mloc = lane & 15, kh = lane >> 4;
  v8f acc = zero8();

  const int aro = tid >> 3, aco = (tid & 7) * 8;    // A: 16 rows x 8 uint4 (tid<128)
  const int bro = tid >> 1, bco = (tid & 1) * 32;   // B: 128 rows x 2 x (4 uint4)

  uint4 ar, br[4];
  auto gload = [&](int k0){
    if(tid < 128) ar = *(const uint4*)&A[(long)aro * K + k0 + aco];
#pragma unroll
    for(int j = 0; j < 4; j++) br[j] = *(const uint4*)&Bm[(long)(n0 + bro) * K + k0 + bco + j * 8];
  };
  auto lstore = [&](int buf){
    if(tid < 128) *(uint4*)&Al[buf][aro][aco] = ar;
#pragma unroll
    for(int j = 0; j < 4; j++) *(uint4*)&Bl[buf][bro][bco + j * 8] = br[j];
  };
  auto gasync = [&](int k0, int buf){
    if(tid < 128) async_b128(&A[(long)aro * K + k0 + aco], &Al[buf][aro][aco]);
#pragma unroll
    for(int j = 0; j < 4; j++)
      async_b128(&Bm[(long)(n0 + bro) * K + k0 + bco + j * 8], &Bl[buf][bro][bco + j * 8]);
  };
  auto compute = [&](int buf){
#pragma unroll
    for(int q = 0; q < 2; q++){
      FragB af, bfm;
      af.q[0]  = *(const uint4*)&Al[buf][mloc][q * 32 + kh * 16];
      af.q[1]  = *(const uint4*)&Al[buf][mloc][q * 32 + kh * 16 + 8];
      int cc   = w * 16 + mloc;
      bfm.q[0] = *(const uint4*)&Bl[buf][cc][q * 32 + kh * 16];
      bfm.q[1] = *(const uint4*)&Bl[buf][cc][q * 32 + kh * 16 + 8];
      acc = __builtin_amdgcn_wmma_f32_16x16x32_bf16(
          false, af.v, false, bfm.v, (short)0, acc, false, false);
    }
  };

  const int nk = K >> 6;
#if HAVE_ASYNC_LDS
  gasync(0, 0);
  wait_async0();
  __syncthreads();
  for(int kt = 0; kt < nk; kt++){
    int buf = kt & 1;
    if(kt + 1 < nk) gasync((kt + 1) * 64, buf ^ 1);
    compute(buf);
    wait_async0();
    __syncthreads();
  }
#else
  gload(0);
  lstore(0);
  __syncthreads();
  for(int kt = 0; kt < nk; kt++){
    int buf = kt & 1;
    if(kt + 1 < nk) gload((kt + 1) * 64);
    compute(buf);
    if(kt + 1 < nk) lstore(buf ^ 1);
    __syncthreads();
  }
#endif

  // stage f32 C tile [16][128] in LDS (overlay on Bl), coalesced b128 stores
  float* Cs = (float*)&Bl[0][0][0];
#pragma unroll
  for(int r = 0; r < 8; r++)
    Cs[(r + 8 * kh) * 128 + w * 16 + mloc] = alpha * acc[r];
  __syncthreads();
  const int row = tid >> 4, seg = tid & 15;
#pragma unroll
  for(int jj = 0; jj < 2; jj++){
    long idx = (long)b * 16 * N + (long)row * N + n0 + (seg * 2 + jj) * 4;
    *(uint4*)&Cf[idx] = *(uint4*)&Cs[row * 128 + (seg * 2 + jj) * 4];
  }
}

// ---------------------------------------------------------------------------
// Slot-axis softmax over i (16), +EPS, in place. One thread per (b, j).
// ---------------------------------------------------------------------------
__global__ void k_softmax_col(float* __restrict__ dots){
  int g = blockIdx.x * 256 + threadIdx.x;       // 32 * 1024
  if(g >= 32 * 1024) return;
  int b = g >> 10, j = g & 1023;
  float* p = dots + (long)b * 16 * 1024 + j;
  float mx = -1e30f;
#pragma unroll
  for(int i = 0; i < 16; i++) mx = fmaxf(mx, p[i * 1024]);
  float e[16]; float s = 0.f;
#pragma unroll
  for(int i = 0; i < 16; i++){ e[i] = expf(p[i * 1024] - mx); s += e[i]; }
  float inv = 1.f / s;
#pragma unroll
  for(int i = 0; i < 16; i++) p[i * 1024] = e[i] * inv + 1e-8f;
}

__global__ void k_rowsum(const float* __restrict__ attn, float* __restrict__ rsum){
  int row = blockIdx.x;   // 512
  const float* p = attn + (long)row * 1024;
  __shared__ float s[256];
  float l = 0.f;
  for(int i = threadIdx.x; i < 1024; i += 256) l += p[i];
  s[threadIdx.x] = l;
  __syncthreads();
  for(int st = 128; st > 0; st >>= 1){
    if(threadIdx.x < st) s[threadIdx.x] += s[threadIdx.x + st];
    __syncthreads();
  }
  if(threadIdx.x == 0) rsum[row] = s[0];
}

__global__ void k_attn_norm(const float* __restrict__ attn, const float* __restrict__ rsum,
                            u16* __restrict__ attnb, float* __restrict__ amap, int write_map){
  long g = (long)blockIdx.x * 256 + threadIdx.x;   // 512 * 1024
  if(g >= 512L * 1024) return;
  int row = (int)(g >> 10), j = (int)(g & 1023);
  float v = attn[g] / rsum[row];
  attnb[g] = f2bf(v);
  if(write_map){
    int b = row >> 4, i = row & 15;
    amap[(long)b * 1024 * 16 + (long)j * 16 + i] = v;
  }
}

// ---------------------------------------------------------------------------
// Tiny encoder self-attention: 16 tokens, 8 heads, dh=64. One block per (b,h).
// ---------------------------------------------------------------------------
__global__ __launch_bounds__(256)
void k_enc_attn(const u16* __restrict__ qa, const u16* __restrict__ ka,
                const u16* __restrict__ va, u16* __restrict__ ob){
  int blk = blockIdx.x; int b = blk >> 3, h = blk & 7;
  int tid = threadIdx.x; int i = tid >> 4, j = tid & 15;
  __shared__ float S[16][17];
  const long rowbase = (long)b * 16;
  const int  cb = h * 64;

  const u16* qr = qa + (rowbase + i) * 512 + cb;
  const u16* kr = ka + (rowbase + j) * 512 + cb;
  float s = 0.f;
  for(int d = 0; d < 64; d++) s += bf2f(qr[d]) * bf2f(kr[d]);
  S[i][j] = s * 0.125f;
  __syncthreads();
  float mx = -1e30f;
  for(int t = 0; t < 16; t++) mx = fmaxf(mx, S[i][t]);
  float e = expf(S[i][j] - mx);
  __syncthreads();
  S[i][j] = e;
  __syncthreads();
  float sum = 0.f;
  for(int t = 0; t < 16; t++) sum += S[i][t];
  float p = e / sum;
  __syncthreads();
  S[i][j] = p;
  __syncthreads();
  for(int dd = 0; dd < 4; dd++){
    int d = j * 4 + dd;
    float o = 0.f;
    for(int t = 0; t < 16; t++) o += S[i][t] * bf2f(va[(rowbase + t) * 512 + cb + d]);
    ob[(rowbase + i) * 512 + cb + d] = f2bf(o);
  }
}

__global__ void k_glu(const float* __restrict__ ug, u16* __restrict__ tb){
  long g = (long)blockIdx.x * 256 + threadIdx.x;
  if(g >= 512L * 3072) return;
  long row = g / 3072; int c = (int)(g % 3072);
  float u  = ug[row * 6144 + c];
  float gv = ug[row * 6144 + 3072 + c];
  float si = gv / (1.f + expf(-gv));
  tb[g] = f2bf(u * si);
}

// ---------------------------------------------------------------------------
extern "C" void kernel_launch(void* const* d_in, const int* in_sizes, int n_in,
                              void* d_out, int out_size, void* d_ws, size_t ws_size,
                              hipStream_t stream){
  (void)in_sizes; (void)n_in; (void)out_size; (void)ws_size;
  const float* x       = (const float*)d_in[0];
  const float* noise   = (const float*)d_in[1];
  const float* init_mu = (const float*)d_in[2];
  const float* init_ls = (const float*)d_in[3];
  const float* Wk  = (const float*)d_in[4];
  const float* Wv  = (const float*)d_in[5];
  const float* Wq  = (const float*)d_in[6];
  const float* ni_g = (const float*)d_in[7],  *ni_b = (const float*)d_in[8];
  const float* ns_g = (const float*)d_in[9],  *ns_b = (const float*)d_in[10];
  const float* nica_g = (const float*)d_in[11], *nica_b = (const float*)d_in[12];
  const float* ln1_g = (const float*)d_in[13], *ln1_b = (const float*)d_in[14];
  const float* Wqa = (const float*)d_in[15], *Wka = (const float*)d_in[16];
  const float* Wva = (const float*)d_in[17], *Woa = (const float*)d_in[18];
  const float* ln2_g = (const float*)d_in[19], *ln2_b = (const float*)d_in[20];
  const float* W1 = (const float*)d_in[21], *W2 = (const float*)d_in[22];
  const float* lnf_g = (const float*)d_in[23], *lnf_b = (const float*)d_in[24];

  const int Dm = 768, XR = 32768, SR = 512, HD = 512;
  const float SCALE = 1.0f / sqrtf(768.0f);

  char* wsb = (char*)d_ws;
  size_t off = 0;
  auto alloc = [&](size_t bytes) -> char* {
    char* p = wsb + off;
    off = (off + bytes + 255) & ~(size_t)255;
    return p;
  };

  // transposed bf16 weights: WT[N][K]
  u16* wkT  = (u16*)alloc((size_t)768 * 768 * 2);
  u16* wvT  = (u16*)alloc((size_t)768 * 768 * 2);
  u16* wqT  = (u16*)alloc((size_t)768 * 768 * 2);
  u16* wqaT = (u16*)alloc((size_t)512 * 768 * 2);
  u16* wkaT = (u16*)alloc((size_t)512 * 768 * 2);
  u16* wvaT = (u16*)alloc((size_t)512 * 768 * 2);
  u16* woaT = (u16*)alloc((size_t)768 * 512 * 2);
  u16* w1T  = (u16*)alloc((size_t)6144 * 768 * 2);
  u16* w2T  = (u16*)alloc((size_t)768 * 3072 * 2);
  u16* xnb  = (u16*)alloc((size_t)XR * 768 * 2);
  u16* kb   = (u16*)alloc((size_t)XR * 768 * 2);        // [b*1024+j][768]
  u16* vT   = (u16*)alloc((size_t)32 * 768 * 1024 * 2); // [b][768][1024]
  float* slots = (float*)alloc((size_t)SR * 768 * 4);
  float* h     = (float*)alloc((size_t)SR * 768 * 4);
  u16* slnb  = (u16*)alloc((size_t)SR * 768 * 2);
  u16* qb    = (u16*)alloc((size_t)SR * 768 * 2);
  float* dots = (float*)alloc((size_t)32 * 16 * 1024 * 4);
  float* rsum = (float*)alloc((size_t)512 * 4);
  u16* attnb = (u16*)alloc((size_t)32 * 16 * 1024 * 2);
  float* upd = (float*)alloc((size_t)SR * 768 * 4);
  u16* ab    = (u16*)alloc((size_t)SR * 768 * 2);
  u16* qab   = (u16*)alloc((size_t)SR * 512 * 2);
  u16* kab   = (u16*)alloc((size_t)SR * 512 * 2);
  u16* vab   = (u16*)alloc((size_t)SR * 512 * 2);
  u16* obf   = (u16*)alloc((size_t)SR * 512 * 2);
  u16* fb    = (u16*)alloc((size_t)SR * 768 * 2);
  float* ug  = (float*)alloc((size_t)SR * 6144 * 4);
  u16* tb    = (u16*)alloc((size_t)SR * 3072 * 2);

  float* amap = (float*)d_out + (size_t)SR * 768;

  // --- weights -> transposed bf16
  k_cvt_transpose<<<dim3(24, 24),  256, 0, stream>>>(Wk,  wkT,  768, 768);
  k_cvt_transpose<<<dim3(24, 24),  256, 0, stream>>>(Wv,  wvT,  768, 768);
  k_cvt_transpose<<<dim3(24, 24),  256, 0, stream>>>(Wq,  wqT,  768, 768);
  k_cvt_transpose<<<dim3(16, 24),  256, 0, stream>>>(Wqa, wqaT, 768, 512);
  k_cvt_transpose<<<dim3(16, 24),  256, 0, stream>>>(Wka, wkaT, 768, 512);
  k_cvt_transpose<<<dim3(16, 24),  256, 0, stream>>>(Wva, wvaT, 768, 512);
  k_cvt_transpose<<<dim3(24, 16),  256, 0, stream>>>(Woa, woaT, 512, 768);
  k_cvt_transpose<<<dim3(192, 24), 256, 0, stream>>>(W1,  w1T,  768, 6144);
  k_cvt_transpose<<<dim3(24, 96),  256, 0, stream>>>(W2,  w2T,  3072, 768);

  // --- xn = LN(x); k = xn@Wk (row-major); vT = (xn@Wv)^T per batch
  k_layernorm<0,1,0><<<XR, 256, 0, stream>>>(x, nullptr, ni_g, ni_b, xnb, nullptr, Dm);
  k_gemm<0><<<dim3(12, 256), 256, 0, stream>>>(xnb, wkT, kb, nullptr, XR, Dm, Dm);
  k_gemm<3><<<dim3(12, 256), 256, 0, stream>>>(xnb, wvT, vT, nullptr, XR, Dm, Dm);

  k_init_slots<<<(SR * 768 + 255) / 256, 256, 0, stream>>>(noise, init_mu, init_ls,
                                                           slots, SR * 768, Dm);

  for(int it = 0; it < 4; it++){
    int last = (it == 3) ? 1 : 0;
    k_layernorm<0,1,0><<<SR, 256, 0, stream>>>(slots, nullptr, ns_g, ns_b, slnb, nullptr, Dm);
    k_gemm<0><<<dim3(12, 4), 256, 0, stream>>>(slnb, wqT, qb, nullptr, SR, Dm, Dm);
    // dots = SCALE * q @ k^T  (B rows = keys, [1024][768])
    k_gemm16<<<dim3(8, 32), 256, 0, stream>>>(qb, kb, dots, 1024, Dm, SCALE);
    k_softmax_col<<<128, 256, 0, stream>>>(dots);
    k_rowsum<<<512, 256, 0, stream>>>(dots, rsum);
    k_attn_norm<<<2048, 256, 0, stream>>>(dots, rsum, attnb, amap, last);
    // updates = attn @ v  (B rows = vT[b][768][1024])
    k_gemm16<<<dim3(6, 32), 256, 0, stream>>>(attnb, vT, upd, Dm, 1024, 1.0f);
    k_layernorm<1,0,1><<<SR, 256, 0, stream>>>(slots, upd, nica_g, nica_b, nullptr, h, Dm);
    k_layernorm<0,1,0><<<SR, 256, 0, stream>>>(h, nullptr, ln1_g, ln1_b, ab, nullptr, Dm);
    k_gemm<0><<<dim3(8, 4), 256, 0, stream>>>(ab, wqaT, qab, nullptr, SR, HD, Dm);
    k_gemm<0><<<dim3(8, 4), 256, 0, stream>>>(ab, wkaT, kab, nullptr, SR, HD, Dm);
    k_gemm<0><<<dim3(8, 4), 256, 0, stream>>>(ab, wvaT, vab, nullptr, SR, HD, Dm);
    k_enc_attn<<<256, 256, 0, stream>>>(qab, kab, vab, obf);
    k_gemm<2><<<dim3(12, 4), 256, 0, stream>>>(obf, woaT, nullptr, h, SR, Dm, HD);
    k_layernorm<0,1,0><<<SR, 256, 0, stream>>>(h, nullptr, ln2_g, ln2_b, fb, nullptr, Dm);
    k_gemm<1><<<dim3(96, 4), 256, 0, stream>>>(fb, w1T, nullptr, ug, SR, 6144, Dm);
    k_glu<<<(SR * 3072 + 255) / 256, 256, 0, stream>>>(ug, tb);
    k_gemm<2><<<dim3(12, 4), 256, 0, stream>>>(tb, w2T, nullptr, h, SR, Dm, 3072);
    k_layernorm<0,0,1><<<SR, 256, 0, stream>>>(h, nullptr, lnf_g, lnf_b, nullptr, slots, Dm);
  }

  (void)hipMemcpyAsync(d_out, slots, (size_t)SR * 768 * 4, hipMemcpyDeviceToDevice, stream);
}
